// Model_60713657696383
// MI455X (gfx1250) — compile-verified, matrix operand order checked by reading, not run on hardware
//
#include <hip/hip_runtime.h>
#include <hip/hip_bf16.h>

// ---------------------------------------------------------------------------
// CDNA5 (gfx1250, wave32) spatial-spectral Mamba block.
// GEMMs: v_wmma_f32_16x16x32_f16, compile-time K, zero-padded operands so all
// fragment loads are unconditional b128; B tiles staged in LDS per block.
// ---------------------------------------------------------------------------

typedef __attribute__((ext_vector_type(16))) _Float16 v16h;
typedef __attribute__((ext_vector_type(8)))  _Float16 v8h;
typedef __attribute__((ext_vector_type(4)))  float    v4f;
typedef __attribute__((ext_vector_type(8)))  float    v8f;

#define T_TOK   18432          // 2*96*96 pixels (tokens for row/col mamba)
#define T_SPE   147456         // 18432 * 8 spectral tokens
#define HH      96
#define WW      96
#define BB      2
#define CC      64

// ------------------------------ small kernels ------------------------------

__global__ void k_zero(float* p, int n) {
  int i = blockIdx.x * blockDim.x + threadIdx.x;
  if (i < n) p[i] = 0.f;
}

__global__ void k_negexp(const float* a, float* o, int n) {
  int i = blockIdx.x * blockDim.x + threadIdx.x;
  if (i < n) o[i] = -__expf(a[i]);
}

// x (B,C,H,W) f32  ->  x_tok (T,64) f16, token order t = ((b*96+h)*96+w)
__global__ void k_xtok(const float* __restrict__ x, _Float16* __restrict__ xt) {
  int i = blockIdx.x * blockDim.x + threadIdx.x;
  if (i >= T_TOK * CC) return;
  int c = i & 63;
  int t = i >> 6;
  int b = t / (HH * WW);
  int r = t - b * (HH * WW);
  int h = r / WW, w = r - h * WW;
  xt[i] = (_Float16)x[(((size_t)b * CC + c) * HH + h) * WW + w];
}

// spectral A: (T8, 32) f16, cols 0..7 = x_tok viewed (T8,8), cols 8..31 = 0
__global__ void k_xspe(const _Float16* __restrict__ xt, _Float16* __restrict__ xs) {
  int i = blockIdx.x * blockDim.x + threadIdx.x;
  if (i >= T_SPE * 32) return;
  int g = i & 31;
  int t8 = i >> 5;
  xs[i] = (g < 8) ? xt[t8 * 8 + g] : (_Float16)0.f;
}

// weight prep: f32 [Nreal][K] -> f16 [Npad][Kpad], zero padded
__global__ void k_wpad(const float* __restrict__ W, _Float16* __restrict__ Wp,
                       int Nreal, int K, int Npad, int Kpad) {
  int i = blockIdx.x * blockDim.x + threadIdx.x;
  if (i >= Npad * Kpad) return;
  int n = i / Kpad, k = i - n * Kpad;
  Wp[i] = (n < Nreal && k < K) ? (_Float16)W[n * K + k] : (_Float16)0.f;
}

// ------------------------------- WMMA GEMM ---------------------------------
// Block = 4 waves, each wave one 16(M) x 64(N) tile; K = KT*32, fully unrolled.
// Fragment layout (wave32 16x16x32 f16): element e, lane l:
//   k = 16*(e>>3) + 8*(l>>4) + (e&7),  m|n = l&15
// -> per lane two contiguous 8-element K runs => b128 loads (global or ds).
// B block (64 x KT*32 f16) is one contiguous span of Wp; staged to LDS once.

__device__ __forceinline__ v16h cat8(v8h lo, v8h hi) {
  return __builtin_shufflevector(lo, hi, 0, 1, 2, 3, 4, 5, 6, 7,
                                 8, 9, 10, 11, 12, 13, 14, 15);
}

__device__ __forceinline__ v8h cvt8(v4f a, v4f b) {
  v8h r;
  r[0] = (_Float16)a[0]; r[1] = (_Float16)a[1];
  r[2] = (_Float16)a[2]; r[3] = (_Float16)a[3];
  r[4] = (_Float16)b[0]; r[5] = (_Float16)b[1];
  r[6] = (_Float16)b[2]; r[7] = (_Float16)b[3];
  return r;
}

__device__ __forceinline__ float silu_f(float z) {
  return z / (1.f + __expf(-z));
}

template <int KT, bool AF16, bool GATED, int EMODE>
__global__ void k_gemm(const void* __restrict__ Av, const float* __restrict__ Zg,
                       const _Float16* __restrict__ Wp, float* __restrict__ Cd,
                       int M, int Nreal, int lda, int ldz, int ldc,
                       int L, int phase, int dir, float scale) {
  __shared__ _Float16 bsh[64 * KT * 32];     // <= 16 KB of the 320 KB WGP LDS

  const int lane = threadIdx.x & 31;
  const int wave = threadIdx.x >> 5;
  const int n0 = blockIdx.y * 64;

  // ---- cooperative B-block stage: contiguous span, b128 copies ----
  {
    const v8h* src = (const v8h*)(Wp + (size_t)n0 * (KT * 32));
    v8h* dst = (v8h*)bsh;
#pragma unroll
    for (int i = 0; i < KT * 2; ++i)
      dst[threadIdx.x + i * 128] = src[threadIdx.x + i * 128];
  }
  __syncthreads();

  const int m0 = (blockIdx.x * 4 + wave) * 16;
  if (m0 >= M) return;                       // never taken (M % 64 == 0)
  const int kb = (lane >> 4) << 3;           // 0 or 8
  const int mrow = m0 + (lane & 15);
  const int nc = lane & 15;

  v8f acc[4];
  const v8f vz = {0.f, 0.f, 0.f, 0.f, 0.f, 0.f, 0.f, 0.f};
#pragma unroll
  for (int i = 0; i < 4; ++i) acc[i] = vz;

#pragma unroll
  for (int kt = 0; kt < KT; ++kt) {
    const int kbase = kt * 32 + kb;
    v16h af;
    if (AF16) {
      const _Float16* pa = (const _Float16*)Av + (size_t)mrow * lda + kbase;
      af = cat8(*(const v8h*)pa, *(const v8h*)(pa + 16));
    } else {
      const float* pa = (const float*)Av + (size_t)mrow * lda + kbase;
      v4f a0 = *(const v4f*)pa,        a1 = *(const v4f*)(pa + 4);
      v4f a2 = *(const v4f*)(pa + 16), a3 = *(const v4f*)(pa + 20);
      if (GATED) {
        const float* pz = Zg + (size_t)mrow * ldz + kbase;
        v4f z0 = *(const v4f*)pz,        z1 = *(const v4f*)(pz + 4);
        v4f z2 = *(const v4f*)(pz + 16), z3 = *(const v4f*)(pz + 20);
#pragma unroll
        for (int j = 0; j < 4; ++j) {
          a0[j] *= silu_f(z0[j]); a1[j] *= silu_f(z1[j]);
          a2[j] *= silu_f(z2[j]); a3[j] *= silu_f(z3[j]);
        }
      }
      af = cat8(cvt8(a0, a1), cvt8(a2, a3));
    }
#pragma unroll
    for (int nt = 0; nt < 4; ++nt) {
      const _Float16* pb = bsh + (nt * 16 + nc) * (KT * 32) + kbase;
      v16h bf = cat8(*(const v8h*)pb, *(const v8h*)(pb + 16));
      acc[nt] = __builtin_amdgcn_wmma_f32_16x16x32_f16(
          false, af, false, bf, (short)0, acc[nt], false, false);
    }
  }

  // ---- epilogue ----
  const int mbase = m0 + ((lane >> 4) << 3);
#pragma unroll
  for (int nt = 0; nt < 4; ++nt) {
    const int n = n0 + nt * 16 + nc;
    if (n >= Nreal) continue;
#pragma unroll
    for (int r = 0; r < 8; ++r) {
      const int m = mbase + r;
      const float v = acc[nt][r];
      if (EMODE == 0) {
        Cd[(size_t)m * ldc + n] = v;
      } else {
        // sequence-local token m -> original token t, with bidir flip
        int nseq = m / L, l = m - nseq * L;
        int lo = dir ? (L - 1 - l) : l;
        long t;
        if (phase == 1) {                    // column sequences: n=(b,w)
          int b = nseq / WW, w = nseq - b * WW;
          t = (long)b * (HH * WW) + (long)lo * WW + w;
        } else {                             // row / linear sequences
          t = (long)nseq * L + lo;
        }
        Cd[t * 64 + n] += scale * v;
      }
    }
  }
}

// --------------------- depthwise causal conv (w=4) + SiLU ------------------
__global__ void k_conv_silu(const float* __restrict__ xz,
                            const float* __restrict__ cw,
                            const float* __restrict__ cb,
                            float* __restrict__ u,
                            int nseq, int L, int di, int ldxz, int ldu,
                            int phase, int dir) {
  int idx = blockIdx.x * blockDim.x + threadIdx.x;
  if (idx >= nseq * L * di) return;
  int d  = idx % di;
  int tl = idx / di;
  int l  = tl % L, n = tl / L;
  float s = cb[d];
#pragma unroll
  for (int j = 0; j < 4; ++j) {
    int m = l - 3 + j;
    if (m < 0) continue;
    int lo = dir ? (L - 1 - m) : m;
    long t;
    if (phase == 1) { int b = n / WW, w = n - b * WW; t = (long)b * (HH * WW) + (long)lo * WW + w; }
    else            { t = (long)n * L + lo; }
    s += cw[d * 4 + j] * xz[t * (long)ldxz + d];
  }
  u[(size_t)tl * ldu + d] = s / (1.f + __expf(-s));
}

// -------------------- dt = softplus(dtin @ W_dt^T + b_dt) ------------------
__global__ void k_dt(const float* __restrict__ xdbl,
                     const float* __restrict__ Wdt,
                     const float* __restrict__ bdt,
                     float* __restrict__ dt,
                     int T, int di, int dtr, int ldx) {
  int idx = blockIdx.x * blockDim.x + threadIdx.x;
  if (idx >= T * di) return;
  int d = idx % di, t = idx / di;
  float s = bdt[d];
  for (int r = 0; r < dtr; ++r) s += xdbl[(size_t)t * ldx + r] * Wdt[d * dtr + r];
  dt[idx] = (s > 20.f) ? s : log1pf(__expf(s));
}

// --------------------------- selective scan --------------------------------
__global__ void k_scan(const float* __restrict__ dt,
                       const float* __restrict__ u,
                       const float* __restrict__ xdbl,
                       const float* __restrict__ Amat,
                       const float* __restrict__ Dv,
                       float* __restrict__ y,
                       int nseq, int L, int di, int dtr, int ldx,
                       int ldu, int ldy) {
  int idx = blockIdx.x * blockDim.x + threadIdx.x;
  if (idx >= nseq * di) return;
  int d = idx % di, n = idx / di;
  float A[16], h[16];
#pragma unroll
  for (int s = 0; s < 16; ++s) { A[s] = Amat[d * 16 + s]; h[s] = 0.f; }
  const float Dd = Dv[d];
  for (int l = 0; l < L; ++l) {
    size_t tl = (size_t)n * L + l;
    float dtv = dt[tl * di + d];
    float uv  = u[tl * ldu + d];
    const float* bc = xdbl + tl * ldx + dtr;   // B at [0..16), C at [16..32)
    float du = dtv * uv, acc = 0.f;
#pragma unroll
    for (int s = 0; s < 16; ++s) {
      h[s] = h[s] * __expf(dtv * A[s]) + du * bc[s];
      acc += h[s] * bc[16 + s];
    }
    y[tl * ldy + d] = acc + uv * Dd;
  }
}

// --------------------------- group-norm stats ------------------------------
__global__ void k_gnstats(const float* __restrict__ spa,
                          const float* __restrict__ spe,
                          float* __restrict__ mu, float* __restrict__ rstd) {
  const int bid = blockIdx.x;
  const float* src = (bid < 8) ? spa : spe;
  int rem = bid & 7, b = rem >> 2, gm = rem & 3;
  const int elems = (HH * WW) * 16;
  float s = 0.f, ss = 0.f;
  for (int i = threadIdx.x; i < elems; i += blockDim.x) {
    int t = b * (HH * WW) + (i >> 4);
    int c = gm * 16 + (i & 15);
    float v = src[(size_t)t * 64 + c];
    s += v; ss += v * v;
  }
  __shared__ float sh_s[256], sh_q[256];
  sh_s[threadIdx.x] = s; sh_q[threadIdx.x] = ss;
  __syncthreads();
  for (int st = 128; st > 0; st >>= 1) {
    if ((int)threadIdx.x < st) {
      sh_s[threadIdx.x] += sh_s[threadIdx.x + st];
      sh_q[threadIdx.x] += sh_q[threadIdx.x + st];
    }
    __syncthreads();
  }
  if (threadIdx.x == 0) {
    float m = sh_s[0] / elems;
    float var = sh_q[0] / elems - m * m;
    mu[bid] = m;
    rstd[bid] = rsqrtf(var + 1e-5f);
  }
}

// ------------------------------ final combine ------------------------------
__global__ void k_combine(const float* __restrict__ spa,
                          const float* __restrict__ spe,
                          const float* __restrict__ mu,
                          const float* __restrict__ rstd,
                          const float* __restrict__ wsa, const float* __restrict__ bsa,
                          const float* __restrict__ wse, const float* __restrict__ bse,
                          const float* __restrict__ aw,
                          const float* __restrict__ x, float* __restrict__ out) {
  size_t idx = (size_t)blockIdx.x * blockDim.x + threadIdx.x;
  if (idx >= (size_t)BB * CC * HH * WW) return;
  int w = idx % WW;
  int h = (idx / WW) % HH;
  int c = (idx / (HH * WW)) % CC;
  int b = idx / ((size_t)CC * HH * WW);
  size_t t = ((size_t)b * HH + h) * WW + w;
  int si = b * 4 + (c >> 4);
  float v1 = (spa[t * 64 + c] - mu[si]) * rstd[si] * wsa[c] + bsa[c];
  v1 = v1 / (1.f + __expf(-v1));
  float v2 = (spe[t * 64 + c] - mu[8 + si]) * rstd[8 + si] * wse[c] + bse[c];
  v2 = v2 / (1.f + __expf(-v2));
  float e0 = __expf(aw[0]), e1 = __expf(aw[1]);
  float w0 = e0 / (e0 + e1);
  out[idx] = v1 * w0 + v2 * (1.f - w0) + x[idx];
}

// ------------------------------- host side ---------------------------------

extern "C" void kernel_launch(void* const* d_in, const int* in_sizes, int n_in,
                              void* d_out, int out_size, void* d_ws, size_t ws_size,
                              hipStream_t stream) {
  (void)in_sizes; (void)n_in; (void)out_size; (void)ws_size;
  const float* x = (const float*)d_in[0];
  struct P {
    const float *Win, *convw, *convb, *Wx, *Wdt, *bdt, *Alog, *Dv, *Wout;
  };
  auto getP = [&](int base) {
    P p;
    p.Win  = (const float*)d_in[base + 0];
    p.convw= (const float*)d_in[base + 1];
    p.convb= (const float*)d_in[base + 2];
    p.Wx   = (const float*)d_in[base + 3];
    p.Wdt  = (const float*)d_in[base + 4];
    p.bdt  = (const float*)d_in[base + 5];
    p.Alog = (const float*)d_in[base + 6];
    p.Dv   = (const float*)d_in[base + 7];
    p.Wout = (const float*)d_in[base + 8];
    return p;
  };
  P pr = getP(1), pc = getP(10), ps = getP(19);
  const float* gw_spa = (const float*)d_in[28];
  const float* gb_spa = (const float*)d_in[29];
  const float* gw_spe = (const float*)d_in[30];
  const float* gb_spe = (const float*)d_in[31];
  const float* aw     = (const float*)d_in[32];

  // workspace carve-out (256B aligned)
  char* wsb = (char*)d_ws;
  size_t off = 0;
  auto alloc = [&](size_t bytes) {
    void* p = wsb + off;
    off = (off + bytes + 255) & ~(size_t)255;
    return p;
  };
  _Float16* x_tok = (_Float16*)alloc((size_t)T_TOK * 64 * 2);
  _Float16* x_spe = (_Float16*)alloc((size_t)T_SPE * 32 * 2);
  // padded f16 weights
  _Float16* WinRp  = (_Float16*)alloc(256 * 64 * 2);
  _Float16* WinCp  = (_Float16*)alloc(256 * 64 * 2);
  _Float16* WxRp   = (_Float16*)alloc(64 * 128 * 2);
  _Float16* WxCp   = (_Float16*)alloc(64 * 128 * 2);
  _Float16* WoutRp = (_Float16*)alloc(64 * 128 * 2);
  _Float16* WoutCp = (_Float16*)alloc(64 * 128 * 2);
  _Float16* WinSp  = (_Float16*)alloc(64 * 32 * 2);
  _Float16* WxSp   = (_Float16*)alloc(64 * 32 * 2);
  _Float16* WoutSp = (_Float16*)alloc(64 * 32 * 2);
  float* Arow   = (float*)alloc(128 * 16 * 4);
  float* Acol   = (float*)alloc(128 * 16 * 4);
  float* Aspe   = (float*)alloc(16 * 16 * 4);
  float* mu     = (float*)alloc(16 * 4);
  float* rstd   = (float*)alloc(16 * 4);
  float* xz     = (float*)alloc(((size_t)T_TOK * 256 + 64) * 4); // == T_SPE*32 (+guard)
  float* u      = (float*)alloc((size_t)T_SPE * 32 * 4);         // >= T_TOK*128
  float* xdbl   = (float*)alloc((size_t)T_SPE * 33 * 4);         // >= T_TOK*36
  float* dtb    = (float*)alloc((size_t)T_TOK * 128 * 4);        // == T_SPE*16
  float* ybuf   = (float*)alloc((size_t)T_SPE * 32 * 4);         // >= T_TOK*128
  float* accspa = (float*)alloc((size_t)T_TOK * 64 * 4);
  float* ys_spe = (float*)alloc((size_t)T_TOK * 64 * 4);         // viewed (T_SPE, 8)

  const int TPB = 256;
  auto blocks = [&](size_t n) { return (unsigned)((n + TPB - 1) / TPB); };

  // ---- prep ----
  k_xtok<<<blocks((size_t)T_TOK * 64), TPB, 0, stream>>>(x, x_tok);
  k_xspe<<<blocks((size_t)T_SPE * 32), TPB, 0, stream>>>(x_tok, x_spe);
  k_negexp<<<blocks(2048), TPB, 0, stream>>>(pr.Alog, Arow, 2048);
  k_negexp<<<blocks(2048), TPB, 0, stream>>>(pc.Alog, Acol, 2048);
  k_negexp<<<blocks(256), TPB, 0, stream>>>(ps.Alog, Aspe, 256);
  k_wpad<<<blocks(256 * 64), TPB, 0, stream>>>(pr.Win,  WinRp,  256, 64, 256, 64);
  k_wpad<<<blocks(256 * 64), TPB, 0, stream>>>(pc.Win,  WinCp,  256, 64, 256, 64);
  k_wpad<<<blocks(64 * 128), TPB, 0, stream>>>(pr.Wx,   WxRp,    36, 128, 64, 128);
  k_wpad<<<blocks(64 * 128), TPB, 0, stream>>>(pc.Wx,   WxCp,    36, 128, 64, 128);
  k_wpad<<<blocks(64 * 128), TPB, 0, stream>>>(pr.Wout, WoutRp,  64, 128, 64, 128);
  k_wpad<<<blocks(64 * 128), TPB, 0, stream>>>(pc.Wout, WoutCp,  64, 128, 64, 128);
  k_wpad<<<blocks(64 * 32), TPB, 0, stream>>>(ps.Win,  WinSp,   32,   8, 64, 32);
  k_wpad<<<blocks(64 * 32), TPB, 0, stream>>>(ps.Wx,   WxSp,    33,  16, 64, 32);
  k_wpad<<<blocks(64 * 32), TPB, 0, stream>>>(ps.Wout, WoutSp,   8,  16, 64, 32);
  k_zero<<<blocks((size_t)T_TOK * 64), TPB, 0, stream>>>(accspa, T_TOK * 64);

  const dim3 gT(T_TOK / 64);   // 288 blocks of 4 waves (1152 M-tiles)
  const dim3 gS(T_SPE / 64);   // 2304 blocks (9216 M-tiles)

  // ---- row (phase 0) and column (phase 1) bidirectional mamba ----
  for (int phase = 0; phase < 2; ++phase) {
    const P& p = phase ? pc : pr;
    const float* Am = phase ? Acol : Arow;
    const _Float16* Winp  = phase ? WinCp  : WinRp;
    const _Float16* Wxp   = phase ? WxCp   : WxRp;
    const _Float16* Woutp = phase ? WoutCp : WoutRp;
    // in-projection: xz = tokens @ W_in^T (T x 256); token-order invariant
    k_gemm<2, true, false, 0><<<dim3(T_TOK / 64, 4), 128, 0, stream>>>(
        x_tok, nullptr, Winp, xz, T_TOK, 256, 64, 0, 256, 0, 0, 0, 0.f);
    for (int dir = 0; dir < 2; ++dir) {
      k_conv_silu<<<blocks((size_t)192 * 96 * 128), TPB, 0, stream>>>(
          xz, p.convw, p.convb, u, 192, 96, 128, 256, 128, phase, dir);
      k_gemm<4, false, false, 0><<<gT, 128, 0, stream>>>(
          u, nullptr, Wxp, xdbl, T_TOK, 36, 128, 0, 36, 0, 0, 0, 0.f);
      k_dt<<<blocks((size_t)T_TOK * 128), TPB, 0, stream>>>(
          xdbl, p.Wdt, p.bdt, dtb, T_TOK, 128, 4, 36);
      k_scan<<<blocks((size_t)192 * 128), TPB, 0, stream>>>(
          dtb, u, xdbl, Am, p.Dv, ybuf, 192, 96, 128, 4, 36, 128, 128);
      // gated out-projection, mapped-accumulate (0.25 = bidir * spa average)
      k_gemm<4, false, true, 1><<<gT, 128, 0, stream>>>(
          ybuf, xz + 128, Woutp, accspa, T_TOK, 64, 128, 256, 0, 96, phase, dir,
          0.25f);
    }
  }

  // ---- spectral mamba (18432 sequences of L=8, d_model=8, d_inner=16) ----
  // zero u/ybuf so their K-pad columns (16..31) are 0 for the WMMA loaders
  k_zero<<<blocks((size_t)T_SPE * 32), TPB, 0, stream>>>(u, T_SPE * 32);
  k_zero<<<blocks((size_t)T_SPE * 32), TPB, 0, stream>>>(ybuf, T_SPE * 32);
  k_gemm<1, true, false, 0><<<gS, 128, 0, stream>>>(
      x_spe, nullptr, WinSp, xz, T_SPE, 32, 32, 0, 32, 0, 0, 0, 0.f);
  k_conv_silu<<<blocks((size_t)T_TOK * 8 * 16), TPB, 0, stream>>>(
      xz, ps.convw, ps.convb, u, T_TOK, 8, 16, 32, 32, 2, 0);
  k_gemm<1, false, false, 0><<<gS, 128, 0, stream>>>(
      u, nullptr, WxSp, xdbl, T_SPE, 33, 32, 0, 33, 0, 0, 0, 0.f);
  k_dt<<<blocks((size_t)T_SPE * 16), TPB, 0, stream>>>(
      xdbl, ps.Wdt, ps.bdt, dtb, T_SPE, 16, 1, 33);
  k_scan<<<blocks((size_t)T_TOK * 16), TPB, 0, stream>>>(
      dtb, u, xdbl, Aspe, ps.Dv, ybuf, T_TOK, 8, 16, 1, 33, 32, 32);
  k_gemm<1, false, true, 0><<<gS, 128, 0, stream>>>(
      ybuf, xz + 16, WoutSp, ys_spe, T_SPE, 8, 32, 32, 8, 0, 0, 0, 0.f);

  // ---- group norms + weighted combine + residual ----
  k_gnstats<<<16, 256, 0, stream>>>(accspa, ys_spe, mu, rstd);
  k_combine<<<blocks((size_t)BB * CC * HH * WW), TPB, 0, stream>>>(
      accspa, ys_spe, mu, rstd, gw_spa, gb_spa, gw_spe, gb_spe, aw, x,
      (float*)d_out);
}
